// GPTLanguageModel_3246995276532
// MI455X (gfx1250) — matrix operand order repaired and checked by reading, not compiled, not run on hardware
//
#include <hip/hip_runtime.h>
#include <hip/hip_bf16.h>
#include <math.h>

// Problem constants (match reference)
#define BSZ 2
#define TLEN 1024
#define CDIM 768
#define HN 12
#define LNUM 6
#define VDIM 32000
#define HS 64

// GEMM tiling
#define BM 128
#define BN 64
#define BK 32
#define LDT 40   // GEMM LDS row stride in bf16 elems (80B: 16B aligned, bank-spread)

// Flash-attention tiling
#define QT 128   // query tile rows
#define KT 64    // key tile rows
#define LDA 80   // attention LDS row stride in bf16 elems (160B)

typedef __attribute__((ext_vector_type(16))) __bf16 v16bf;
typedef __attribute__((ext_vector_type(8)))  float  v8f;

struct U8x32 { uint4 lo, hi; };   // 32 bytes == sizeof(v16bf)

__device__ __forceinline__ unsigned short f2bf(float f) {
    union { float f; unsigned int u; } cv; cv.f = f;
    unsigned int u = cv.u;
    u += 0x7FFFu + ((u >> 16) & 1u);   // round-to-nearest-even
    return (unsigned short)(u >> 16);
}

// A-fragment gather (16x32 bf16): e0..15 -> k=+0..15 (contiguous)
__device__ __forceinline__ v16bf ld_frag_a(const unsigned short* p) {
    U8x32 r; r.lo = *(const uint4*)p; r.hi = *(const uint4*)(p + 8);
    return __builtin_bit_cast(v16bf, r);
}
// B-fragment gather (32x16 bf16): e0..15 -> k=+0..15 (contiguous)
__device__ __forceinline__ v16bf ld_frag_b(const unsigned short* p) {
    U8x32 r; r.lo = *(const uint4*)p; r.hi = *(const uint4*)(p + 8);
    return __builtin_bit_cast(v16bf, r);
}
__device__ __forceinline__ v8f wmma_bf16(v16bf a, v16bf b, v8f c) {
    return __builtin_amdgcn_wmma_f32_16x16x32_bf16(false, a, false, b,
                                                   (short)0, c, false, false);
}
__device__ __forceinline__ void async_cp16(unsigned int ldsAddr, const void* g) {
    asm volatile("global_load_async_to_lds_b128 %0, %1, off"
                 :: "v"(ldsAddr), "v"((unsigned long long)(size_t)g) : "memory");
}
__device__ __forceinline__ void wait_async0() {
    asm volatile("s_wait_asynccnt 0x0" ::: "memory");
}
__device__ __forceinline__ void wait_ds0() {
    asm volatile("s_wait_dscnt 0x0" ::: "memory");
}

// ---------------------------------------------------------------------------
// fp32 -> bf16 elementwise convert (weights; one pass per matrix per call)
// ---------------------------------------------------------------------------
__global__ void cvt_kernel(const float* __restrict__ src,
                           unsigned short* __restrict__ dst, int n) {
    int i = blockIdx.x * blockDim.x + threadIdx.x;
    if (i < n) dst[i] = f2bf(src[i]);
}

// ---------------------------------------------------------------------------
// Embedding: x[b,t,c] = tok_emb[idx[b,t], c] + pos_emb[t, c]   (fp32 out)
// ---------------------------------------------------------------------------
__global__ void embed_kernel(const int* __restrict__ idx,
                             const float* __restrict__ tok,
                             const float* __restrict__ pos,
                             float* __restrict__ x) {
    int i = blockIdx.x * blockDim.x + threadIdx.x;
    const int total = BSZ * TLEN * CDIM;
    if (i >= total) return;
    int c  = i % CDIM;
    int bt = i / CDIM;
    int t  = bt % TLEN;
    int tk = idx[bt];
    x[i] = tok[(size_t)tk * CDIM + c] + pos[(size_t)t * CDIM + c];
}

// ---------------------------------------------------------------------------
// LayerNorm over last dim (C=768), one block (256 thr) per row; bf16 output
// ---------------------------------------------------------------------------
__global__ void ln_kernel(const float* __restrict__ x,
                          const float* __restrict__ g,
                          const float* __restrict__ b,
                          unsigned short* __restrict__ out) {
    __shared__ float s1[256], s2[256];
    const int row = blockIdx.x;
    const int tid = threadIdx.x;
    const float* xr = x + (size_t)row * CDIM;
    float sum = 0.f, sq = 0.f;
    for (int c = tid; c < CDIM; c += 256) { float v = xr[c]; sum += v; sq += v * v; }
    s1[tid] = sum; s2[tid] = sq;
    __syncthreads();
    for (int off = 128; off > 0; off >>= 1) {
        if (tid < off) { s1[tid] += s1[tid + off]; s2[tid] += s2[tid + off]; }
        __syncthreads();
    }
    const float mean = s1[0] * (1.0f / CDIM);
    const float var  = s2[0] * (1.0f / CDIM) - mean * mean;
    const float inv  = rsqrtf(var + 1e-5f);
    unsigned short* outr = out + (size_t)row * CDIM;
    for (int c = tid; c < CDIM; c += 256)
        outr[c] = f2bf((xr[c] - mean) * inv * g[c] + b[c]);
}

// ---------------------------------------------------------------------------
// bf16-WMMA GEMM: C[z] = A (MxK, bf16) @ B[z] (KxN, bf16)
//   fused: +bias[n], +residual (fp32, C layout), ReLU; fp32 and/or bf16 out.
// Block tile 128x64, K step 32, 8 waves, each wave owns 32x32 -> 4 WMMA/step.
// A tile via GLOBAL_LOAD_ASYNC_TO_LDS_B128; B tile transposed into LDS [n][k].
// ---------------------------------------------------------------------------
__global__ void gemm_wmma_kernel(const unsigned short* __restrict__ A,
                                 const unsigned short* __restrict__ Bm,
                                 int K, int lda, int ldb, int ldc,
                                 long sB, long sC,
                                 float* __restrict__ outF,
                                 unsigned short* __restrict__ outH,
                                 const float* __restrict__ bias,
                                 const float* __restrict__ res,
                                 int relu) {
    __shared__ __align__(16) unsigned short As[BM * LDT];   // [m][k]
    __shared__ __align__(16) unsigned short Bs[BN * LDT];   // [n][k] (transposed)

    const int tid = threadIdx.x;
    const int bm  = blockIdx.y * BM;
    const int bn  = blockIdx.x * BN;
    const int z   = blockIdx.z;
    const unsigned short* Bb = Bm + (size_t)z * sB;

    const int lane = tid & 31;
    const int wave = tid >> 5;
    const int wm   = wave & 3;        // 32-row granule
    const int wn   = wave >> 2;       // 32-col granule
    const int fr   = lane & 15;
    const int sel  = lane >> 4;

    const unsigned short* aRow0 = As + (wm * 32 + fr) * LDT + sel * 16;
    const unsigned short* aRow1 = As + (wm * 32 + 16 + fr) * LDT + sel * 16;
    const unsigned short* bCol0 = Bs + (wn * 32 + fr) * LDT + sel * 16;
    const unsigned short* bCol1 = Bs + (wn * 32 + 16 + fr) * LDT + sel * 16;

    const int rowA = tid >> 2;            // 0..63 (and +64)
    const int chA  = tid & 3;
    const unsigned int ldsA0 = (unsigned int)(size_t)&As[rowA * LDT + chA * 8];
    const unsigned int ldsA1 = (unsigned int)(size_t)&As[(rowA + 64) * LDT + chA * 8];
    const unsigned short* gA0 = A + (size_t)(bm + rowA) * lda + chA * 8;
    const unsigned short* gA1 = A + (size_t)(bm + rowA + 64) * lda + chA * 8;

    const int kB  = tid & 31;
    const int nB  = (tid >> 5) * 8;
    const unsigned short* gB = Bb + (size_t)kB * ldb + bn + nB;

    v8f zero = {0.f,0.f,0.f,0.f,0.f,0.f,0.f,0.f};
    v8f acc00 = zero, acc01 = zero, acc10 = zero, acc11 = zero;

    for (int kt = 0; kt < K; kt += BK) {
        async_cp16(ldsA0, gA0 + kt);
        async_cp16(ldsA1, gA1 + kt);
        {
            const unsigned short* p = gB + (size_t)kt * ldb;
            union { uint4 v; unsigned short s[8]; } u;
            u.v = *(const uint4*)p;
#pragma unroll
            for (int j = 0; j < 8; ++j) Bs[(nB + j) * LDT + kB] = u.s[j];
            if (kt + BK < K)
                __builtin_prefetch(p + (size_t)BK * ldb, 0, 1);
        }
        wait_async0();
        __syncthreads();

        v16bf a0 = ld_frag_a(aRow0);
        v16bf a1 = ld_frag_a(aRow1);
        v16bf b0 = ld_frag_b(bCol0);
        v16bf b1 = ld_frag_b(bCol1);

        acc00 = wmma_bf16(a0, b0, acc00);
        acc01 = wmma_bf16(a0, b1, acc01);
        acc10 = wmma_bf16(a1, b0, acc10);
        acc11 = wmma_bf16(a1, b1, acc11);
        __syncthreads();
    }

    const int col0 = bn + wn * 32 + fr;
#pragma unroll
    for (int ai = 0; ai < 2; ++ai) {
        const int rowBase = bm + wm * 32 + ai * 16 + sel * 8;
        const v8f* accL = ai ? &acc10 : &acc00;
        const v8f* accR = ai ? &acc11 : &acc01;
#pragma unroll
        for (int r = 0; r < 8; ++r) {
            const int row = rowBase + r;
            const size_t i0 = (size_t)row * ldc + col0;
            const size_t i1 = i0 + 16;
            float v0 = (*accL)[r], v1 = (*accR)[r];
            if (bias) { v0 += bias[col0]; v1 += bias[col0 + 16]; }
            if (res)  { v0 += res[i0]; v1 += res[i1]; }
            if (relu) { v0 = fmaxf(v0, 0.f); v1 = fmaxf(v1, 0.f); }
            if (outF) {
                outF[(size_t)z * sC + i0] = v0;
                outF[(size_t)z * sC + i1] = v1;
            }
            if (outH) {
                outH[(size_t)z * sC + i0] = f2bf(v0);
                outH[(size_t)z * sC + i1] = f2bf(v1);
            }
        }
    }
}

// ---------------------------------------------------------------------------
// WMMA flash attention, causal. One block (256 thr, 8 waves) per
// (b, h, 128-query tile). q/k/v: bf16, layout [(h*B+b)][t][s].
// Each wave owns 16 query rows x 64 cols. Per 64-key tile:
//   S = Q Kt (8 WMMA, K's [t][s] layout IS the B-frag [n][k] layout),
//   online softmax in registers (row stats per lane-half, shfl_xor reduce),
//   P (bf16) -> LDS (wave-private rows: only s_wait_dscnt, no barrier),
//   O += P V (8 WMMA, V transposed into LDS).
// Output bf16 head-concat: out[(b*T+t)*C + h*HS + s]
// ---------------------------------------------------------------------------
__global__ void fattn_kernel(const unsigned short* __restrict__ q,
                             const unsigned short* __restrict__ k,
                             const unsigned short* __restrict__ vv,
                             unsigned short* __restrict__ out, float scale) {
    __shared__ __align__(16) unsigned short Qs[QT * LDA];
    __shared__ __align__(16) unsigned short Ks[KT * LDA];
    __shared__ __align__(16) unsigned short Vt[HS * LDA];
    __shared__ __align__(16) unsigned short Ps[QT * LDA];

    const int tid = threadIdx.x;
    const int qt  = blockIdx.x;
    const int h   = blockIdx.y;
    const int b   = blockIdx.z;
    const int tq0 = qt * QT;
    const size_t base = ((size_t)(h * BSZ + b)) * TLEN * HS;

    const int lane = tid & 31;
    const int wid  = tid >> 5;
    const int fr   = lane & 15;
    const int sel  = lane >> 4;

    // Q tile (128x64 bf16 = 1024 16B chunks) via async copy
#pragma unroll
    for (int i = 0; i < 4; ++i) {
        const int c = tid + 256 * i;
        const int row = c >> 3, ch = c & 7;
        async_cp16((unsigned int)(size_t)&Qs[row * LDA + ch * 8],
                   q + base + (size_t)(tq0 + row) * HS + ch * 8);
    }

    float m_run[8], l_run[8];
#pragma unroll
    for (int r = 0; r < 8; ++r) { m_run[r] = -3.0e38f; l_run[r] = 0.f; }
    v8f zero = {0.f,0.f,0.f,0.f,0.f,0.f,0.f,0.f};
    v8f accO[4];
#pragma unroll
    for (int c = 0; c < 4; ++c) accO[c] = zero;

    const unsigned short* aQ = Qs + (wid * 16 + fr) * LDA + sel * 16;
    const unsigned short* aP = Ps + (wid * 16 + fr) * LDA + sel * 16;
    const int rloc = wid * 16 + sel * 8;   // this lane's first local row

    const int jmax = (tq0 + QT - 1) / KT;  // inclusive causal bound
    for (int j = 0; j <= jmax; ++j) {
        const int tk0 = j * KT;
        // K tile: async copy, natural [t][s] layout
#pragma unroll
        for (int i = 0; i < 2; ++i) {
            const int c = tid + 256 * i;
            const int row = c >> 3, ch = c & 7;
            async_cp16((unsigned int)(size_t)&Ks[row * LDA + ch * 8],
                       k + base + (size_t)(tk0 + row) * HS + ch * 8);
        }
        // V tile: transpose-scatter into Vt[s][t]
#pragma unroll
        for (int i = 0; i < 2; ++i) {
            const int c = tid + 256 * i;
            const int t = c >> 3, s0 = (c & 7) * 8;
            union { uint4 v; unsigned short s[8]; } u;
            u.v = *(const uint4*)(vv + base + (size_t)(tk0 + t) * HS + s0);
#pragma unroll
            for (int e = 0; e < 8; ++e) Vt[(s0 + e) * LDA + t] = u.s[e];
        }
        wait_async0();
        __syncthreads();

        // S = Q Kt : wave's 16 rows x 64 cols, 2 K-steps x 4 col tiles
        v8f accS[4];
#pragma unroll
        for (int c = 0; c < 4; ++c) accS[c] = zero;
#pragma unroll
        for (int kk = 0; kk < HS; kk += 32) {
            v16bf a = ld_frag_a(aQ + kk);
#pragma unroll
            for (int c = 0; c < 4; ++c) {
                v16bf bb = ld_frag_b(Ks + (c * 16 + fr) * LDA + kk + sel * 16);
                accS[c] = wmma_bf16(a, bb, accS[c]);
            }
        }

        // online softmax on the 16x64 chunk (row stats wave-local)
#pragma unroll
        for (int r = 0; r < 8; ++r) {
            const int row = tq0 + rloc + r;
            float sv[4];
#pragma unroll
            for (int c = 0; c < 4; ++c) {
                const int col = tk0 + c * 16 + fr;
                const float s = accS[c][r] * scale;
                sv[c] = (col > row) ? -3.0e38f : s;
            }
            float tm = fmaxf(fmaxf(sv[0], sv[1]), fmaxf(sv[2], sv[3]));
#pragma unroll
            for (int m = 1; m < 16; m <<= 1)
                tm = fmaxf(tm, __shfl_xor(tm, m, 32));
            const float nm   = fmaxf(m_run[r], tm);
            const float corr = __expf(m_run[r] - nm);
            m_run[r] = nm;
            float ps = 0.f;
#pragma unroll
            for (int c = 0; c < 4; ++c) {
                const float p = __expf(sv[c] - nm);
                ps += p;
                Ps[(rloc + r) * LDA + c * 16 + fr] = f2bf(p);
            }
#pragma unroll
            for (int m = 1; m < 16; m <<= 1)
                ps += __shfl_xor(ps, m, 32);
            l_run[r] = l_run[r] * corr + ps;
#pragma unroll
            for (int c = 0; c < 4; ++c) accO[c][r] = accO[c][r] * corr;
        }
        // P rows are wave-private: in-order LDS per wave, just drain DScnt
        wait_ds0();

        // O += P V
#pragma unroll
        for (int kk = 0; kk < KT; kk += 32) {
            v16bf a = ld_frag_a(aP + kk);
#pragma unroll
            for (int c = 0; c < 4; ++c) {
                v16bf bb = ld_frag_b(Vt + (c * 16 + fr) * LDA + kk + sel * 16);
                accO[c] = wmma_bf16(a, bb, accO[c]);
            }
        }
        __syncthreads();   // protect Ks/Vt before next tile's loads
    }

    // write O / l  (bf16, head-concat layout)
#pragma unroll
    for (int r = 0; r < 8; ++r) {
        const int t = tq0 + rloc + r;
        const float inv = 1.f / l_run[r];
#pragma unroll
        for (int c = 0; c < 4; ++c)
            out[((size_t)(b * TLEN + t)) * CDIM + h * HS + c * 16 + fr]
                = f2bf(accO[c][r] * inv);
    }
}

// ---------------------------------------------------------------------------
// Host orchestration
// ---------------------------------------------------------------------------
static inline size_t align256(size_t x) { return (x + 255) & ~(size_t)255; }

extern "C" void kernel_launch(void* const* d_in, const int* in_sizes, int n_in,
                              void* d_out, int out_size, void* d_ws, size_t ws_size,
                              hipStream_t stream) {
    (void)in_sizes; (void)n_in; (void)out_size; (void)ws_size;

    const int*   idx  = (const int*)  d_in[0];
    const float* tok  = (const float*)d_in[1];
    const float* pos  = (const float*)d_in[2];
    const float* Wq   = (const float*)d_in[3];
    const float* Wk   = (const float*)d_in[4];
    const float* Wv   = (const float*)d_in[5];
    const float* Wp   = (const float*)d_in[6];
    const float* bp   = (const float*)d_in[7];
    const float* ln1g = (const float*)d_in[8];
    const float* ln1b = (const float*)d_in[9];
    const float* ln2g = (const float*)d_in[10];
    const float* ln2b = (const float*)d_in[11];
    const float* W1   = (const float*)d_in[12];
    const float* b1   = (const float*)d_in[13];
    const float* W2   = (const float*)d_in[14];
    const float* b2   = (const float*)d_in[15];
    const float* lnfg = (const float*)d_in[16];
    const float* lnfb = (const float*)d_in[17];
    const float* Wh   = (const float*)d_in[18];
    const float* bh   = (const float*)d_in[19];

    const int M = BSZ * TLEN;               // 2048 rows
    const int nQKV = HN * CDIM * HS;        // 589824
    const int nWp  = CDIM * CDIM;           // 589824
    const int nW1  = CDIM * 4 * CDIM;       // 2359296
    const int nWh  = CDIM * VDIM;           // 24576000

    char* wsb = (char*)d_ws;
    size_t off = 0;
    float* x = (float*)(wsb + off);                      off = align256(off + (size_t)M * CDIM * 4);
    unsigned short* hlnH = (unsigned short*)(wsb + off); off = align256(off + (size_t)M * CDIM * 2);
    unsigned short* qbH  = (unsigned short*)(wsb + off); off = align256(off + (size_t)HN * M * HS * 2);
    unsigned short* kbH  = (unsigned short*)(wsb + off); off = align256(off + (size_t)HN * M * HS * 2);
    unsigned short* vbH  = (unsigned short*)(wsb + off); off = align256(off + (size_t)HN * M * HS * 2);
    unsigned short* abH  = (unsigned short*)(wsb + off); off = align256(off + (size_t)M * CDIM * 2);
    unsigned short* mlpH = (unsigned short*)(wsb + off); off = align256(off + (size_t)M * 4 * CDIM * 2);
    unsigned short* wbuf = (unsigned short*)(wsb + off); off = align256(off + (size_t)nWh * 2);

    unsigned short* wqH = wbuf;
    unsigned short* wkH = wqH + nQKV;
    unsigned short* wvH = wkH + nQKV;
    unsigned short* wpH = wvH + nQKV;
    unsigned short* w1H = wpH + nWp;
    unsigned short* w2H = w1H + nW1;

    embed_kernel<<<(M * CDIM + 255) / 256, 256, 0, stream>>>(idx, tok, pos, x);

    const float scale = 1.0f / sqrtf((float)CDIM);  // reference scales by C^-0.5
    const long  wstride = (long)CDIM * HS;
    const long  qstride = (long)M * HS;

    for (int l = 0; l < LNUM; ++l) {
        cvt_kernel<<<(nQKV + 255) / 256, 256, 0, stream>>>(Wq + (size_t)l * nQKV, wqH, nQKV);
        cvt_kernel<<<(nQKV + 255) / 256, 256, 0, stream>>>(Wk + (size_t)l * nQKV, wkH, nQKV);
        cvt_kernel<<<(nQKV + 255) / 256, 256, 0, stream>>>(Wv + (size_t)l * nQKV, wvH, nQKV);
        cvt_kernel<<<(nWp  + 255) / 256, 256, 0, stream>>>(Wp + (size_t)l * nWp,  wpH, nWp);
        cvt_kernel<<<(nW1  + 255) / 256, 256, 0, stream>>>(W1 + (size_t)l * nW1,  w1H, nW1);
        cvt_kernel<<<(nW1  + 255) / 256, 256, 0, stream>>>(W2 + (size_t)l * nW1,  w2H, nW1);

        ln_kernel<<<M, 256, 0, stream>>>(x, ln1g + (size_t)l * CDIM, ln1b + (size_t)l * CDIM, hlnH);

        // Q/K/V: batched over heads, bf16 outputs for WMMA attention
        dim3 gqkv(HS / BN, M / BM, HN);
        gemm_wmma_kernel<<<gqkv, 256, 0, stream>>>(hlnH, wqH, CDIM, CDIM, HS, HS,
            wstride, qstride, nullptr, qbH, nullptr, nullptr, 0);
        gemm_wmma_kernel<<<gqkv, 256, 0, stream>>>(hlnH, wkH, CDIM, CDIM, HS, HS,
            wstride, qstride, nullptr, kbH, nullptr, nullptr, 0);
        gemm_wmma_kernel<<<gqkv, 256, 0, stream>>>(hlnH, wvH, CDIM, CDIM, HS, HS,
            wstride, qstride, nullptr, vbH, nullptr, nullptr, 0);

        // WMMA flash attention -> abH
        fattn_kernel<<<dim3(TLEN / QT, HN, BSZ), 256, 0, stream>>>(qbH, kbH, vbH, abH, scale);

        // x = x + abH @ Wp + bp
        gemm_wmma_kernel<<<dim3(CDIM / BN, M / BM, 1), 256, 0, stream>>>(abH, wpH,
            CDIM, CDIM, CDIM, CDIM, 0, 0, x, nullptr, bp + (size_t)l * CDIM, x, 0);

        ln_kernel<<<M, 256, 0, stream>>>(x, ln2g + (size_t)l * CDIM, ln2b + (size_t)l * CDIM, hlnH);

        // mlpH = relu(h2 @ W1 + b1)
        gemm_wmma_kernel<<<dim3(4 * CDIM / BN, M / BM, 1), 256, 0, stream>>>(hlnH, w1H,
            CDIM, CDIM, 4 * CDIM, 4 * CDIM, 0, 0, nullptr, mlpH,
            b1 + (size_t)l * 4 * CDIM, nullptr, 1);

        // x = x + mlpH @ W2 + b2
        gemm_wmma_kernel<<<dim3(CDIM / BN, M / BM, 1), 256, 0, stream>>>(mlpH, w2H,
            4 * CDIM, 4 * CDIM, CDIM, CDIM, 0, 0, x, nullptr,
            b2 + (size_t)l * CDIM, x, 0);
    }

    // logits = LN_f(x) @ Wh + bh
    ln_kernel<<<M, 256, 0, stream>>>(x, lnfg, lnfb, hlnH);
    cvt_kernel<<<(nWh + 255) / 256, 256, 0, stream>>>(Wh, wbuf, nWh);
    gemm_wmma_kernel<<<dim3(VDIM / BN, M / BM, 1), 256, 0, stream>>>(hlnH, wbuf,
        CDIM, CDIM, VDIM, VDIM, 0, 0, (float*)d_out, nullptr, bh, nullptr, 0);
}
